// NIUcausal_multi_head_self_attention_41214506172840
// MI455X (gfx1250) — matrix operand order repaired and checked by reading, not compiled, not run on hardware
//
#include <hip/hip_runtime.h>

#define DM 1024
#define NH 16
#define HD 64
#define BB 2
#define SS 2048
#define MTOT (BB * SS)  // 4096 rows

typedef __attribute__((ext_vector_type(16))) __bf16       v16bf;
typedef __attribute__((ext_vector_type(8)))  float        v8f;
typedef __attribute__((ext_vector_type(4)))  unsigned int v4u;
typedef __attribute__((ext_vector_type(4)))  unsigned int u32x4;
typedef __attribute__((ext_vector_type(8)))  int          i32x8;
typedef __attribute__((ext_vector_type(4)))  int          i32x4;

union FragB { v16bf v; v4u q[2]; };

__device__ __forceinline__ unsigned short f2bf(float f) {
  union { float f; unsigned int u; } c; c.f = f;
  unsigned int u = c.u + 0x7FFFu + ((c.u >> 16) & 1u);  // round-to-nearest-even
  return (unsigned short)(u >> 16);
}
__device__ __forceinline__ float bf2f(unsigned short b) {
  union { unsigned int u; float f; } c; c.u = ((unsigned int)b) << 16;
  return c.f;
}

// XOR butterfly reductions across 16-lane groups via ds_swizzle_b32
// (offset = {and_mask=0x1f, or_mask=0, xor_mask=k} -> 0x7C00|k, offset[15]=0)
__device__ __forceinline__ float fmax_xor16(float x) {
  x = fmaxf(x, __int_as_float(__builtin_amdgcn_ds_swizzle(__float_as_int(x), 0x7C01)));
  x = fmaxf(x, __int_as_float(__builtin_amdgcn_ds_swizzle(__float_as_int(x), 0x7C02)));
  x = fmaxf(x, __int_as_float(__builtin_amdgcn_ds_swizzle(__float_as_int(x), 0x7C04)));
  x = fmaxf(x, __int_as_float(__builtin_amdgcn_ds_swizzle(__float_as_int(x), 0x7C08)));
  return x;
}
__device__ __forceinline__ float fsum_xor16(float x) {
  x += __int_as_float(__builtin_amdgcn_ds_swizzle(__float_as_int(x), 0x7C01));
  x += __int_as_float(__builtin_amdgcn_ds_swizzle(__float_as_int(x), 0x7C02));
  x += __int_as_float(__builtin_amdgcn_ds_swizzle(__float_as_int(x), 0x7C04));
  x += __int_as_float(__builtin_amdgcn_ds_swizzle(__float_as_int(x), 0x7C08));
  return x;
}

// ---------------------------------------------------------------- TDM
// 2D tile load via Tensor Data Mover, per CDNA5 D# layout:
// group0 = {count, lds_addr, global_addr, type=2},
// group1 = {flags/pad, tensor_dim0/1, tile_dim0/1, tensor_dim0_stride}.
// w0pack carries data_size=2B, pad_enable and pad_interval/amount so the
// TDM itself writes the bank-conflict-padded LDS layout.
__device__ __forceinline__ void tdm_load_tile(unsigned lds_addr,
                                              unsigned long long gaddr,
                                              unsigned w0pack,
                                              unsigned tensor_dim0,
                                              unsigned tensor_dim1,
                                              unsigned tile_dim0,
                                              unsigned tile_dim1,
                                              unsigned long long stride0) {
  u32x4 g0;
  g0.x = 1u;                                   // count=1, user descriptor
  g0.y = lds_addr;                             // LDS byte address
  g0.z = (unsigned)(gaddr & 0xFFFFFFFFull);    // global_addr[31:0]
  g0.w = (unsigned)((gaddr >> 32) & 0x1FFFFFFull) | (2u << 30);  // [56:32] | type=2
  i32x8 g1;
  g1[0] = (int)w0pack;                                        // mask/ds/pad flags
  g1[1] = (int)((tensor_dim0 & 0xFFFFu) << 16);               // abar=0 | tdim0[15:0]
  g1[2] = (int)(((tensor_dim0 >> 16) & 0xFFFFu) | ((tensor_dim1 & 0xFFFFu) << 16));
  g1[3] = (int)(((tensor_dim1 >> 16) & 0xFFFFu) | ((tile_dim0 & 0xFFFFu) << 16));
  g1[4] = (int)(tile_dim1 & 0xFFFFu);                         // tile_dim2 = 0
  g1[5] = (int)(stride0 & 0xFFFFFFFFull);                     // dim0_stride[31:0]
  g1[6] = (int)((stride0 >> 32) & 0xFFFFull);                 // dim0_stride[47:32]
  g1[7] = 0;
  i32x4 gz = {0, 0, 0, 0};
#if defined(__clang_major__) && (__clang_major__ >= 23)
  i32x8 gz8 = {0, 0, 0, 0, 0, 0, 0, 0};
  __builtin_amdgcn_tensor_load_to_lds(g0, g1, gz, gz, gz8, 0);
#else
  __builtin_amdgcn_tensor_load_to_lds(g0, g1, gz, gz, 0);
#endif
}

// ---------------------------------------------------------------- convert
__global__ void cvt_f32_to_bf16(const float* __restrict__ in,
                                unsigned short* __restrict__ out, int n) {
  int i = blockIdx.x * blockDim.x + threadIdx.x;
  if (i < n) out[i] = f2bf(in[i]);
}

// ---------------------------------------------------------------- GEMM
// C[M,N] = A[M,K] * W[N,K]^T  (A, W bf16, K-contiguous).  Block = 128
// threads (4 waves), 128x64 C tile (32 M-rows per wave -> 8 WMMAs per
// k-step against one LDS weight tile).  Weight tile staged by TDM into
// double-buffered LDS (pad feature -> stride 40 halfwords), pipelined with
// s_wait_tensorcnt(1).
template <bool OUT_BF16>
__global__ void __launch_bounds__(128, 1)
gemm_bf16(const unsigned short* __restrict__ A,
          const unsigned short* __restrict__ W,
          void* __restrict__ C, int M, int N, int K) {
  __shared__ unsigned short Wt[2][64 * 40];
  const int lane = threadIdx.x & 31;
  const int wave = threadIdx.x >> 5;       // 0..3
  const int m0   = blockIdx.x * 128 + wave * 32;
  const int n0   = blockIdx.y * 64;
  const int row  = lane & 15;              // operand row/col within 16
  const int sel  = (lane >> 4) * 8;        // K-chunk select per ISA A/B layout

  // data_size=2B(code1), pad_enable, pad_interval=16 DW(code3), pad_amount=4 DW(code3)
  const unsigned w0pack = (1u << 16) | (1u << 20) | (3u << 22) | (3u << 25);

  v8f acc0[4] = {};   // rows m0..m0+15
  v8f acc1[4] = {};   // rows m0+16..m0+31

  if (wave == 0) {  // wave-uniform branch: TDM issued once (EXEC ignored)
    tdm_load_tile((unsigned)(size_t)&Wt[0][0],
                  (unsigned long long)(W + (size_t)n0 * K),
                  w0pack, (unsigned)K, 1u << 20, 32u, 64u,
                  (unsigned long long)K);
  }

  int buf = 0;
  for (int k0 = 0; k0 < K; k0 += 32) {
    if (wave == 0) {
      if (k0 + 32 < K) {
        tdm_load_tile((unsigned)(size_t)&Wt[buf ^ 1][0],
                      (unsigned long long)(W + (size_t)n0 * K + k0 + 32),
                      w0pack, (unsigned)K, 1u << 20, 32u, 64u,
                      (unsigned long long)K);
        __builtin_amdgcn_s_wait_tensorcnt(1);  // in-order: oldest tile landed
      } else {
        __builtin_amdgcn_s_wait_tensorcnt(0);
      }
    }
    __syncthreads();

    // A fragments: lane m holds row m%16, k = {sel..sel+7, 16+sel..16+sel+7}
    FragB af0, af1;
    const unsigned short* ap0 = A + (size_t)(m0 + row) * K + k0 + sel;
    const unsigned short* ap1 = A + (size_t)(m0 + 16 + row) * K + k0 + sel;
    af0.q[0] = *(const v4u*)(ap0);
    af0.q[1] = *(const v4u*)(ap0 + 16);
    af1.q[0] = *(const v4u*)(ap1);
    af1.q[1] = *(const v4u*)(ap1 + 16);
    __builtin_prefetch(ap0 + 32, 0, 0);  // next k-slice -> global_prefetch_b8
    __builtin_prefetch(ap1 + 32, 0, 0);

    for (int nt = 0; nt < 4; ++nt) {
      FragB bf;
      const unsigned short* bp = &Wt[buf][(nt * 16 + row) * 40 + sel];
      bf.q[0] = *(const v4u*)(bp);
      bf.q[1] = *(const v4u*)(bp + 16);
      acc0[nt] = __builtin_amdgcn_wmma_f32_16x16x32_bf16(
          false, af0.v, false, bf.v, (short)0, acc0[nt], false, false);
      acc1[nt] = __builtin_amdgcn_wmma_f32_16x16x32_bf16(
          false, af1.v, false, bf.v, (short)0, acc1[nt], false, false);
    }
    __syncthreads();
    buf ^= 1;
  }

  // C layout: VGPR v, lane l -> (row = v + 8*(l>>4), col = l&15)
  for (int nt = 0; nt < 4; ++nt) {
    int col   = n0 + nt * 16 + row;
    int rbase = m0 + ((lane >> 4) << 3);
    for (int v = 0; v < 8; ++v) {
      size_t i0 = (size_t)(rbase + v) * N + col;
      size_t i1 = (size_t)(rbase + 16 + v) * N + col;
      if (OUT_BF16) {
        ((unsigned short*)C)[i0] = f2bf(acc0[nt][v]);
        ((unsigned short*)C)[i1] = f2bf(acc1[nt][v]);
      } else {
        ((float*)C)[i0] = acc0[nt][v];
        ((float*)C)[i1] = acc1[nt][v];
      }
    }
  }
}

// ---------------------------------------------------------------- RoPE + head split
// in : bf16 [B*S, DM]   (row = b*S+s, col = h*HD+d)
// out: bf16 [B*NH, S, HD]
__global__ void rope_split(const unsigned short* __restrict__ in,
                           unsigned short* __restrict__ out,
                           const int* __restrict__ pos, int apply) {
  int idx = blockIdx.x * blockDim.x + threadIdx.x;  // one even/odd pair each
  const int total = BB * SS * NH * (HD / 2);
  if (idx >= total) return;
  int p = idx & 31; int t = idx >> 5;   // pair index within head
  int h = t & (NH - 1); t >>= 4;
  int s = t & (SS - 1); t >>= 11;
  int b = t;
  const unsigned short* src = in + ((size_t)(b * SS + s)) * DM + h * HD + p * 2;
  float x1 = bf2f(src[0]), x2 = bf2f(src[1]);
  float oe, oo;
  if (apply) {
    float inv = __expf(-((float)(2 * p) / (float)HD) * __logf(10000.0f));
    float ang = (float)pos[s] * inv;
    float sn, cs; __sincosf(ang, &sn, &cs);
    oe = x1 * cs - x2 * sn;
    oo = x1 * sn + x2 * cs;
  } else { oe = x1; oo = x2; }
  unsigned short* dst = out + ((size_t)(b * NH + h) * SS + s) * HD + p * 2;
  dst[0] = f2bf(oe); dst[1] = f2bf(oo);
}

// ---------------------------------------------------------------- flash attention
// One wave (32 threads) per 16-query tile; stream 32-key blocks.
// Q and K tiles staged by TDM (pad feature -> stride 72); Q fragments are
// re-read from LDS each iteration instead of living in VGPRs (kills the
// scratch spill seen in round 2).  V transposed via lanes into LDS.
// Q/K/V: bf16 [B*NH, S, HD].  O: bf16 [B, S, DM] (heads recombined).
__global__ void __launch_bounds__(32, 1)
flash_attn(const unsigned short* __restrict__ Q,
           const unsigned short* __restrict__ Km,
           const unsigned short* __restrict__ Vm,
           unsigned short* __restrict__ O) {
  __shared__ unsigned short Qt[16 * 72];  // [q][d]     TDM-padded stride 72
  __shared__ unsigned short Kt[32 * 72];  // [key][d]   TDM-padded stride 72
  __shared__ unsigned short Vt[64 * 40];  // [d][key]   transposed, stride 40
  __shared__ unsigned short Pt[16 * 40];  // [q][key]   stride 40
  const int lane  = threadIdx.x;
  const int qb    = blockIdx.x;
  const int h     = blockIdx.y, b = blockIdx.z;
  const int bh    = b * NH + h;
  const int qbase = qb * 16;
  const int row   = lane & 15;
  const int sel   = (lane >> 4) * 8;
  const float scale = 0.125f;             // 1/sqrt(64)

  // data_size=2B, pad_enable, pad_interval=32 DW(code4), pad_amount=4 DW(code3)
  const unsigned w0k = (1u << 16) | (1u << 20) | (4u << 22) | (3u << 25);

  // ---- Q tile 16x64 via TDM, loaded once (in-order with K TDMs)
  tdm_load_tile((unsigned)(size_t)Qt,
                (unsigned long long)(Q + ((size_t)bh * SS + qbase) * HD),
                w0k, (unsigned)HD, 1u << 20, (unsigned)HD, 16u,
                (unsigned long long)HD);

  v8f o0 = {}, o1 = {}, o2 = {}, o3 = {};
  float mrow[8], lrow[8];
  for (int v = 0; v < 8; ++v) { mrow[v] = -3.0e38f; lrow[v] = 0.f; }

  const int nkb = (qbase + 15) / 32 + 1;
  for (int kb = 0; kb < nkb; ++kb) {
    const int kbase = kb * 32;

    // ---- K tile 32x64 via TDM (single wave -> issued once)
    tdm_load_tile((unsigned)(size_t)Kt,
                  (unsigned long long)(Km + ((size_t)bh * SS + kbase) * HD),
                  w0k, (unsigned)HD, 1u << 20, (unsigned)HD, 32u,
                  (unsigned long long)HD);

    // ---- V tile: lane owns key row (kbase+lane), transpose into [d][key]
    {
      const unsigned int* vp =
          (const unsigned int*)(Vm + ((size_t)bh * SS + kbase + lane) * HD);
      for (int j = 0; j < 32; ++j) {
        unsigned int w = vp[j];
        Vt[(2 * j + 0) * 40 + lane] = (unsigned short)(w & 0xffffu);
        Vt[(2 * j + 1) * 40 + lane] = (unsigned short)(w >> 16);
      }
    }
    __builtin_amdgcn_s_wait_tensorcnt(0);  // Q (iter 0) and current K landed
    __syncthreads();

    // ---- scores: s0 = keys 0..15, s1 = keys 16..31 (each K=64 via 2 WMMAs)
    v8f s0 = {}, s1 = {};
    {
      FragB qa0, qa1;
      const unsigned short* qp = &Qt[row * 72 + sel];
      qa0.q[0] = *(const v4u*)(qp);      qa0.q[1] = *(const v4u*)(qp + 16);
      qa1.q[0] = *(const v4u*)(qp + 32); qa1.q[1] = *(const v4u*)(qp + 48);

      FragB kf;
      const unsigned short* k0p = &Kt[row * 72 + sel];
      kf.q[0] = *(const v4u*)(k0p);      kf.q[1] = *(const v4u*)(k0p + 16);
      s0 = __builtin_amdgcn_wmma_f32_16x16x32_bf16(false, qa0.v, false, kf.v, (short)0, s0, false, false);
      kf.q[0] = *(const v4u*)(k0p + 32); kf.q[1] = *(const v4u*)(k0p + 48);
      s0 = __builtin_amdgcn_wmma_f32_16x16x32_bf16(false, qa1.v, false, kf.v, (short)0, s0, false, false);
      const unsigned short* k1p = &Kt[(16 + row) * 72 + sel];
      kf.q[0] = *(const v4u*)(k1p);      kf.q[1] = *(const v4u*)(k1p + 16);
      s1 = __builtin_amdgcn_wmma_f32_16x16x32_bf16(false, qa0.v, false, kf.v, (short)0, s1, false, false);
      kf.q[0] = *(const v4u*)(k1p + 32); kf.q[1] = *(const v4u*)(k1p + 48);
      s1 = __builtin_amdgcn_wmma_f32_16x16x32_bf16(false, qa1.v, false, kf.v, (short)0, s1, false, false);
    }

    // ---- causal mask + online softmax (rows live in 16-lane halves)
    for (int v = 0; v < 8; ++v) {
      const int qg = qbase + v + ((lane >> 4) << 3);
      s0[v] = ((kbase + row)      <= qg) ? s0[v] * scale : -3.0e38f;
      s1[v] = ((kbase + 16 + row) <= qg) ? s1[v] * scale : -3.0e38f;

      const float t     = fmax_xor16(fmaxf(s0[v], s1[v]));
      const float mnew  = fmaxf(mrow[v], t);
      const float alpha = __expf(mrow[v] - mnew);
      mrow[v] = mnew;
      const float p0 = __expf(s0[v] - mnew);
      const float p1 = __expf(s1[v] - mnew);
      const float rs = fsum_xor16(p0 + p1);
      lrow[v] = lrow[v] * alpha + rs;
      s0[v] = p0; s1[v] = p1;
      o0[v] *= alpha; o1[v] *= alpha; o2[v] *= alpha; o3[v] *= alpha;
    }

    // ---- P: C-layout -> A-layout via LDS
    for (int v = 0; v < 8; ++v) {
      const int r = v + ((lane >> 4) << 3);
      Pt[r * 40 + row]      = f2bf(s0[v]);
      Pt[r * 40 + 16 + row] = f2bf(s1[v]);
    }
    __syncthreads();
    FragB pf;
    {
      const unsigned short* pp = &Pt[row * 40 + sel];
      pf.q[0] = *(const v4u*)(pp);
      pf.q[1] = *(const v4u*)(pp + 16);
    }
    // ---- O += P (16x32) * V (32x64) : 4 WMMAs over d-tiles
    {
      FragB vf;
      const unsigned short* v0p = &Vt[(0 * 16 + row) * 40 + sel];
      vf.q[0] = *(const v4u*)(v0p); vf.q[1] = *(const v4u*)(v0p + 16);
      o0 = __builtin_amdgcn_wmma_f32_16x16x32_bf16(false, pf.v, false, vf.v, (short)0, o0, false, false);
      const unsigned short* v1p = &Vt[(1 * 16 + row) * 40 + sel];
      vf.q[0] = *(const v4u*)(v1p); vf.q[1] = *(const v4u*)(v1p + 16);
      o1 = __builtin_amdgcn_wmma_f32_16x16x32_bf16(false, pf.v, false, vf.v, (short)0, o1, false, false);
      const unsigned short* v2p = &Vt[(2 * 16 + row) * 40 + sel];
      vf.q[0] = *(const v4u*)(v2p); vf.q[1] = *(const v4u*)(v2p + 16);
      o2 = __builtin_amdgcn_wmma_f32_16x16x32_bf16(false, pf.v, false, vf.v, (short)0, o2, false, false);
      const unsigned short* v3p = &Vt[(3 * 16 + row) * 40 + sel];
      vf.q[0] = *(const v4u*)(v3p); vf.q[1] = *(const v4u*)(v3p + 16);
      o3 = __builtin_amdgcn_wmma_f32_16x16x32_bf16(false, pf.v, false, vf.v, (short)0, o3, false, false);
    }
    __syncthreads();
  }

  // ---- normalize + store (recombine heads): O[b][s][h*HD + d]
  for (int v = 0; v < 8; ++v) {
    const float inv = 1.0f / lrow[v];
    const int s = qbase + v + ((lane >> 4) << 3);
    unsigned short* op = O + ((size_t)b * SS + s) * DM + h * HD;
    op[0 * 16 + row] = f2bf(o0[v] * inv);
    op[1 * 16 + row] = f2bf(o1[v] * inv);
    op[2 * 16 + row] = f2bf(o2[v] * inv);
    op[3 * 16 + row] = f2bf(o3[v] * inv);
  }
}

// ---------------------------------------------------------------- launch
extern "C" void kernel_launch(void* const* d_in, const int* in_sizes, int n_in,
                              void* d_out, int out_size, void* d_ws, size_t ws_size,
                              hipStream_t stream) {
  const float* x   = (const float*)d_in[0];
  const int*   pos = (const int*)d_in[1];
  const float* Wq  = (const float*)d_in[2];
  const float* Wk  = (const float*)d_in[3];
  const float* Wv  = (const float*)d_in[4];
  const float* Wo  = (const float*)d_in[5];

  char* ws = (char*)d_ws;
  size_t off = 0;
  auto alloc = [&](size_t bytes) -> char* {
    char* p = ws + off; off += (bytes + 255) & ~(size_t)255; return p;
  };
  unsigned short* xb   = (unsigned short*)alloc((size_t)MTOT * DM * 2);
  unsigned short* wqb  = (unsigned short*)alloc((size_t)DM * DM * 2);
  unsigned short* wkb  = (unsigned short*)alloc((size_t)DM * DM * 2);
  unsigned short* wvb  = (unsigned short*)alloc((size_t)DM * DM * 2);
  unsigned short* wob  = (unsigned short*)alloc((size_t)DM * DM * 2);
  unsigned short* qtmp = (unsigned short*)alloc((size_t)MTOT * DM * 2);
  unsigned short* ktmp = (unsigned short*)alloc((size_t)MTOT * DM * 2);
  unsigned short* vtmp = (unsigned short*)alloc((size_t)MTOT * DM * 2);
  unsigned short* qr   = (unsigned short*)alloc((size_t)MTOT * DM * 2);
  unsigned short* kr   = (unsigned short*)alloc((size_t)MTOT * DM * 2);
  unsigned short* vr   = (unsigned short*)alloc((size_t)MTOT * DM * 2);
  unsigned short* ob   = qtmp;  // qtmp dead after rope_split -> reuse for attn out

  const int nx = MTOT * DM;
  const int nw = DM * DM;
  cvt_f32_to_bf16<<<(nx + 255) / 256, 256, 0, stream>>>(x,  xb,  nx);
  cvt_f32_to_bf16<<<(nw + 255) / 256, 256, 0, stream>>>(Wq, wqb, nw);
  cvt_f32_to_bf16<<<(nw + 255) / 256, 256, 0, stream>>>(Wk, wkb, nw);
  cvt_f32_to_bf16<<<(nw + 255) / 256, 256, 0, stream>>>(Wv, wvb, nw);
  cvt_f32_to_bf16<<<(nw + 255) / 256, 256, 0, stream>>>(Wo, wob, nw);

  dim3 gg(MTOT / 128, DM / 64);
  gemm_bf16<true><<<gg, 128, 0, stream>>>(xb, wqb, qtmp, MTOT, DM, DM);
  gemm_bf16<true><<<gg, 128, 0, stream>>>(xb, wkb, ktmp, MTOT, DM, DM);
  gemm_bf16<true><<<gg, 128, 0, stream>>>(xb, wvb, vtmp, MTOT, DM, DM);

  const int np = BB * SS * NH * (HD / 2);
  rope_split<<<(np + 255) / 256, 256, 0, stream>>>(qtmp, qr, pos, 1);
  rope_split<<<(np + 255) / 256, 256, 0, stream>>>(ktmp, kr, pos, 1);
  rope_split<<<(np + 255) / 256, 256, 0, stream>>>(vtmp, vr, pos, 0);

  dim3 ga(SS / 16, NH, BB);
  flash_attn<<<ga, 32, 0, stream>>>(qr, kr, vr, ob);

  gemm_bf16<false><<<gg, 128, 0, stream>>>(ob, wob, (float*)d_out, MTOT, DM, DM);
}
